// Head_41455024341743
// MI455X (gfx1250) — matrix operand order repaired and checked by reading, not compiled, not run on hardware
//
#include <hip/hip_runtime.h>
#include <math.h>

// Problem constants (fixed by the reference)
#define WB   1024
#define WT   256
#define WC   384
#define WHS  64
#define NQKV 192   // 3 * HS

typedef _Float16 h16 __attribute__((ext_vector_type(16)));
typedef _Float16 h8  __attribute__((ext_vector_type(8)));
typedef float    f8  __attribute__((ext_vector_type(8)));
typedef float    f4  __attribute__((ext_vector_type(4)));

static __device__ __forceinline__ f8 wmma_f16(h16 a, h16 b, f8 c) {
  // D = A(16x32 f16) * B(32x16 f16) + C(16x16 f32)
  return __builtin_amdgcn_wmma_f32_16x16x32_f16(
      /*neg_a=*/false, a, /*neg_b=*/false, b,
      /*c_mod=*/(short)0, c, /*reuse_a=*/false, /*reuse_b=*/false);
}

// ---------------------------------------------------------------------------
// Kernel 0: pack Wq|Wk|Wv (each [C][HS] f32) into WhT[192][384] f16,
// WhT[n][k] = W_sel[k][n%64]. Transposed so each WMMA B-fragment
// (lane = output col, 16 contiguous K values) is one 32-byte contiguous load.
// ---------------------------------------------------------------------------
__global__ void build_wht(const float* __restrict__ Wq,
                          const float* __restrict__ Wk,
                          const float* __restrict__ Wv,
                          _Float16* __restrict__ WhT) {
  int i = blockIdx.x * blockDim.x + threadIdx.x;
  if (i >= NQKV * WC) return;
  int n = i / WC;
  int k = i - n * WC;
  float v;
  if (n < 64)       v = Wq[k * WHS + n];
  else if (n < 128) v = Wk[k * WHS + (n - 64)];
  else              v = Wv[k * WHS + (n - 128)];
  WhT[(size_t)n * WC + k] = (_Float16)v;
}

// Store one 16x16 f32 D tile as f16 into q / k / v-transposed.
// nt is a compile-time constant after unrolling, so the range branch folds.
static __device__ __forceinline__ void store_tile(f8 acc, int nt, int b,
                                                  int m0, int ln, int hi,
                                                  _Float16* __restrict__ qh,
                                                  _Float16* __restrict__ kh,
                                                  _Float16* __restrict__ vt) {
  #pragma unroll
  for (int r = 0; r < 8; ++r) {
    const int row = m0 + r + hi * 8;
    _Float16 hv = (_Float16)acc[r];
    if (nt < 4)
      qh[((size_t)b * WT + row) * WHS + nt * 16 + ln] = hv;
    else if (nt < 8)
      kh[((size_t)b * WT + row) * WHS + (nt - 4) * 16 + ln] = hv;
    else
      vt[((size_t)b * WHS + (nt - 8) * 16 + ln) * WT + row] = hv;  // V^T
  }
}

// ---------------------------------------------------------------------------
// Kernel 1: QKV projection. Grid = B*4 blocks x 128 threads (4 waves).
// Each wave computes a 16(M) x 192(N) strip with K=384, WMMA f16->f32.
// x is read once in f32 and converted to f16 in registers (A fragments cached
// across all N-tiles). N-tiles processed in pairs (two independent WMMA
// accumulation chains) with double-buffered B fragments so loads overlap WMMA.
// __launch_bounds__(128, 1) lifts the VGPR cap (~160 regs live) so the
// A-fragment cache does not spill to scratch.
// ---------------------------------------------------------------------------
__global__ void __launch_bounds__(128, 1)
qkv_proj(const float* __restrict__ x,
         const _Float16* __restrict__ WhT,
         _Float16* __restrict__ qh,
         _Float16* __restrict__ kh,
         _Float16* __restrict__ vt) {
  const int wave = threadIdx.x >> 5;
  const int lane = threadIdx.x & 31;
  const int b    = blockIdx.x >> 2;
  const int m0   = ((blockIdx.x & 3) << 6) + (wave << 4);
  const int ln   = lane & 15;
  const int hi   = lane >> 4;

  // A fragments: 12 K-steps of 16x32, f32 -> f16 convert in-register.
  // A layout per lane: elems 0..15 = K (base + hi*16 + 0..15).
  h16 A[12];
  const float* xrow = x + ((size_t)b * WT + (m0 + ln)) * WC;
  #pragma unroll
  for (int ks = 0; ks < 12; ++ks) {
    const float* p = xrow + ks * 32 + hi * 16;
    f4 c0 = *(const f4*)(p);
    f4 c1 = *(const f4*)(p + 4);
    f4 c2 = *(const f4*)(p + 8);
    f4 c3 = *(const f4*)(p + 12);
    h16 a;
    #pragma unroll
    for (int t = 0; t < 4; ++t) {
      a[t]      = (_Float16)c0[t];
      a[4 + t]  = (_Float16)c1[t];
      a[8 + t]  = (_Float16)c2[t];
      a[12 + t] = (_Float16)c3[t];
    }
    A[ks] = a;
  }

  #pragma unroll
  for (int ntp = 0; ntp < 6; ++ntp) {
    const int n0 = ntp * 32;
    f8 acc0 = {}, acc1 = {};
    const _Float16* w0 = WhT + (size_t)(n0 + ln) * WC + hi * 16;
    const _Float16* w1 = WhT + (size_t)(n0 + 16 + ln) * WC + hi * 16;
    // Double-buffered B fragments: prefetch ks+1 while WMMAing ks.
    h16 b0 = *(const h16*)(w0);
    h16 b1 = *(const h16*)(w1);
    #pragma unroll
    for (int ks = 0; ks < 12; ++ks) {
      h16 nb0, nb1;
      if (ks < 11) {
        nb0 = *(const h16*)(w0 + (ks + 1) * 32);
        nb1 = *(const h16*)(w1 + (ks + 1) * 32);
      }
      acc0 = wmma_f16(A[ks], b0, acc0);   // two independent D->C chains
      acc1 = wmma_f16(A[ks], b1, acc1);
      if (ks < 11) { b0 = nb0; b1 = nb1; }
    }
    store_tile(acc0, n0 / 16,     b, m0, ln, hi, qh, kh, vt);
    store_tile(acc1, n0 / 16 + 1, b, m0, ln, hi, qh, kh, vt);
  }
}

// ---------------------------------------------------------------------------
// Kernel 2: causal flash attention. Grid = B*4 blocks x 128 threads (4 waves).
// Each wave owns 16 query rows, iterates 32 keys per step with online softmax.
// All 8 K/V B-fragment loads for a step are issued as one clause up front so
// their latency overlaps the S-WMMAs and the exp/shuffle softmax math.
// P (probabilities) is round-tripped through LDS (f16) to convert the WMMA
// D-layout into the A-layout for the P*V matmul. Same-wave LDS ordering +
// explicit s_wait_dscnt keeps this correct without block barriers (waves have
// different causal trip counts, so __syncthreads would deadlock).
// ---------------------------------------------------------------------------
__global__ void __launch_bounds__(128, 1)
attn(const _Float16* __restrict__ qh,
     const _Float16* __restrict__ kh,
     const _Float16* __restrict__ vt,
     float* __restrict__ out) {
  __shared__ _Float16 plds[4 * 16 * 32];
  const int wave = threadIdx.x >> 5;
  const int lane = threadIdx.x & 31;
  const int b    = blockIdx.x >> 2;
  const int m0   = ((blockIdx.x & 3) << 6) + (wave << 4);
  const int ln   = lane & 15;
  const int hi   = lane >> 4;
  _Float16* P = plds + wave * 512;

  // Q A-fragments: HS=64 -> 2 K-steps of 32
  h16 Aq[2];
  const _Float16* qrow = qh + ((size_t)b * WT + (m0 + ln)) * WHS;
  #pragma unroll
  for (int cs = 0; cs < 2; ++cs) {
    const _Float16* p = qrow + cs * 32 + hi * 16;
    h8 lo = *(const h8*)(p);
    h8 hh = *(const h8*)(p + 8);
    h16 a;
    #pragma unroll
    for (int t = 0; t < 8; ++t) { a[t] = lo[t]; a[8 + t] = hh[t]; }
    Aq[cs] = a;
  }

  float mS[8], lS[8];
  f8 O[4] = {};
  #pragma unroll
  for (int r = 0; r < 8; ++r) { mS[r] = -INFINITY; lS[r] = 0.0f; }

  const int nsteps = (m0 + 47) >> 5;  // keys [0, m0+16) rounded to 32
  for (int j = 0; j < nsteps; ++j) {
    const int k0 = j * 32;
    const _Float16* kbase = kh + ((size_t)b * WT + k0) * WHS;
    const _Float16* vbase = vt + (size_t)b * WHS * WT + k0;

    // ---- issue all K and V fragment loads for this step up front ----
    h16 bk[2][2];
    #pragma unroll
    for (int t = 0; t < 2; ++t)
      #pragma unroll
      for (int cs = 0; cs < 2; ++cs)
        bk[t][cs] = *(const h16*)(kbase + (size_t)(t * 16 + ln) * WHS +
                                  cs * 32 + hi * 16);
    h16 bv[4];
    #pragma unroll
    for (int nt = 0; nt < 4; ++nt)
      bv[nt] = *(const h16*)(vbase + (size_t)(nt * 16 + ln) * WT + hi * 16);

    // prefetch next step's K rows and V columns (global_prefetch_b8)
    if (j + 1 < nsteps) {
      __builtin_prefetch((const void*)(kbase + (size_t)(32 + lane) * WHS), 0, 1);
      __builtin_prefetch((const void*)(vbase + (size_t)lane * WT + 32), 0, 1);
      __builtin_prefetch((const void*)(vbase + (size_t)(lane + 32) * WT + 32), 0, 1);
    }

    // ---- S = (Q * K^T) * hs^-0.5, two 16x16 tiles, causal mask ----
    float p[2][8];
    #pragma unroll
    for (int t = 0; t < 2; ++t) {
      f8 s = {};
      #pragma unroll
      for (int cs = 0; cs < 2; ++cs) s = wmma_f16(Aq[cs], bk[t][cs], s);
      const int key = k0 + t * 16 + ln;
      #pragma unroll
      for (int r = 0; r < 8; ++r) {
        const int row = m0 + r + hi * 8;
        float sv = s[r] * 0.125f;          // 1/sqrt(64)
        if (key > row) sv = -INFINITY;     // causal
        p[t][r] = sv;
      }
    }

    // ---- online softmax: row max / rescale / exp / row sum ----
    #pragma unroll
    for (int r = 0; r < 8; ++r) {
      float v = fmaxf(p[0][r], p[1][r]);
      v = fmaxf(v, __shfl_xor(v, 1, 16));
      v = fmaxf(v, __shfl_xor(v, 2, 16));
      v = fmaxf(v, __shfl_xor(v, 4, 16));
      v = fmaxf(v, __shfl_xor(v, 8, 16));
      const float mnew  = fmaxf(mS[r], v);
      const float alpha = __expf(mS[r] - mnew);
      mS[r] = mnew;
      p[0][r] = __expf(p[0][r] - mnew);
      p[1][r] = __expf(p[1][r] - mnew);
      float rs = p[0][r] + p[1][r];
      rs += __shfl_xor(rs, 1, 16);
      rs += __shfl_xor(rs, 2, 16);
      rs += __shfl_xor(rs, 4, 16);
      rs += __shfl_xor(rs, 8, 16);
      lS[r] = lS[r] * alpha + rs;
      #pragma unroll
      for (int nt = 0; nt < 4; ++nt) O[nt][r] *= alpha;
    }

    // ---- D-layout -> A-layout via LDS (f16) ----
    #pragma unroll
    for (int t = 0; t < 2; ++t)
      #pragma unroll
      for (int r = 0; r < 8; ++r)
        P[(r + hi * 8) * 32 + t * 16 + ln] = (_Float16)p[t][r];
    asm volatile("s_wait_dscnt 0x0" ::: "memory");  // same-wave LDS RAW

    h16 pa;
    {
      const _Float16* pr = P + ln * 32 + hi * 16;
      h8 lo = *(const h8*)(pr);
      h8 hh = *(const h8*)(pr + 8);
      #pragma unroll
      for (int t = 0; t < 8; ++t) { pa[t] = lo[t]; pa[8 + t] = hh[t]; }
    }

    // ---- O += P * V  (V transposed -> contiguous B fragments) ----
    #pragma unroll
    for (int nt = 0; nt < 4; ++nt) O[nt] = wmma_f16(pa, bv[nt], O[nt]);
  }

  // ---- normalize and write out [B][T][HS] f32 ----
  #pragma unroll
  for (int nt = 0; nt < 4; ++nt) {
    const int col = nt * 16 + ln;
    #pragma unroll
    for (int r = 0; r < 8; ++r) {
      const int row = m0 + r + hi * 8;
      out[((size_t)b * WT + row) * WHS + col] = O[nt][r] / lS[r];
    }
  }
}

// ---------------------------------------------------------------------------
extern "C" void kernel_launch(void* const* d_in, const int* in_sizes, int n_in,
                              void* d_out, int out_size, void* d_ws,
                              size_t ws_size, hipStream_t stream) {
  (void)in_sizes; (void)n_in; (void)out_size; (void)ws_size;

  const float* x  = (const float*)d_in[0];
  const float* Wq = (const float*)d_in[1];
  const float* Wk = (const float*)d_in[2];
  const float* Wv = (const float*)d_in[3];
  float* out = (float*)d_out;

  char* ws = (char*)d_ws;
  _Float16* WhT = (_Float16*)ws;                        // 192*384*2 = 147456 B
  _Float16* qh  = (_Float16*)(ws + (256u << 10));       // pad to 256 KB
  _Float16* kh  = qh + (size_t)WB * WT * WHS;           // +32 MB
  _Float16* vt  = kh + (size_t)WB * WT * WHS;           // +32 MB

  build_wht<<<(NQKV * WC + 255) / 256, 256, 0, stream>>>(Wq, Wk, Wv, WhT);
  qkv_proj<<<WB * 4, 128, 0, stream>>>(x, WhT, qh, kh, vt);
  attn<<<WB * 4, 128, 0, stream>>>(qh, kh, vt, out);
}
